// DilatedSelfAttention_2302102471321
// MI455X (gfx1250) — compile-verified, matrix operand order checked by reading, not run on hardware
//
#include <hip/hip_runtime.h>
#include <hip/hip_bf16.h>

typedef __attribute__((ext_vector_type(16))) _Float16 v16h;
typedef __attribute__((ext_vector_type(8)))  _Float16 v8h;
typedef __attribute__((ext_vector_type(8)))  float    v8f;
typedef __attribute__((ext_vector_type(4)))  unsigned int u32x4;
typedef __attribute__((ext_vector_type(8)))  int          i32x8;
typedef __attribute__((ext_vector_type(4)))  int          i32x4;

#define B_ 4
#define N_ 8192
#define C_ 1024
#define H_ 128
#define M_ 1024   // per-segment length w/r (same for every config)

// Load a WMMA 16-bit A-operand (or B-operand via the "rows of B^T" trick)
// from a row-major row pointer, per the CDNA5 16-bit 16x32 per-lane layout:
// lane holds row M = lane%16; halves [kbase..kbase+7] then [kbase+16..kbase+23],
// kbase = (lane<16 ? 0 : 8).
__device__ __forceinline__ v16h load_a16(const _Float16* row, int kbase) {
  v8h lo = *(const v8h*)(row + kbase);
  v8h hi = *(const v8h*)(row + kbase + 16);
  v16h r;
#pragma unroll
  for (int i = 0; i < 8; ++i) { r[i] = lo[i]; r[8 + i] = hi[i]; }
  return r;
}

// ---------------------------------------------------------------------------
// TDM: async DMA of a 2D tile of f16 (tile_x elems per row, `rows` rows,
// row stride `row_stride_elems`) from global memory into LDS. D# packed per
// CDNA5 ISA 8.3/8.4: group0 = {flags, lds_addr, global_addr, type=2},
// group1 = {data_size=2B, tensor dims, tile dims, dim0 stride}. 2D only, so
// groups 2/3 are zero. Completion tracked with TENSORcnt.
// ---------------------------------------------------------------------------
__device__ __forceinline__ void tdm_load_2d_f16(unsigned lds_addr,
                                                const _Float16* gptr,
                                                unsigned tile_x, unsigned rows,
                                                unsigned long long row_stride_elems)
{
  const unsigned long long ga = (unsigned long long)gptr;
  u32x4 g0;
  g0[0] = 1u;                                            // count=1 (valid)
  g0[1] = lds_addr;                                      // lds_addr[31:0]
  g0[2] = (unsigned)ga;                                  // global_addr[31:0]
  g0[3] = (unsigned)((ga >> 32) & 0x1FFFFFFu) | (2u << 30);  // addr[56:32]|type=2
  i32x8 g1;
  g1[0] = (int)(1u << 16);                               // data_size=1 (2 bytes)
  g1[1] = (int)((tile_x & 0xFFFFu) << 16);               // tensor_dim0[15:0]
  g1[2] = (int)(((tile_x >> 16) & 0xFFFFu) |             // tensor_dim0[31:16]
                ((rows & 0xFFFFu) << 16));               // tensor_dim1[15:0]
  g1[3] = (int)(((rows >> 16) & 0xFFFFu) |               // tensor_dim1[31:16]
                ((tile_x & 0xFFFFu) << 16));             // tile_dim0
  g1[4] = (int)(rows & 0xFFFFu);                         // tile_dim1 (tile_dim2=0)
  g1[5] = (int)(unsigned)(row_stride_elems & 0xFFFFFFFFull);  // dim0_stride[31:0]
  g1[6] = (int)(unsigned)((row_stride_elems >> 32) & 0xFFFFull); // stride[47:32]
  g1[7] = 0;
  const i32x4 z4 = {};
#if __clang_major__ >= 23
  const i32x8 z8 = {};
  __builtin_amdgcn_tensor_load_to_lds(g0, g1, z4, z4, z8, 0);
#else
  __builtin_amdgcn_tensor_load_to_lds(g0, g1, z4, z4, 0);
#endif
}

__device__ __forceinline__ unsigned lds_offset_of(const void* p) {
  return (unsigned)(size_t)p;   // generic LDS address truncates to group offset
}

// ---------------------------------------------------------------------------
// QKV projection: [32768 x 1024] (f32) x [1024 x 128] (f32) -> f16, via WMMA.
// Grid: (rows/128, 3). 256 threads = 8 waves; wave w owns rows w*16..w*16+15.
// ---------------------------------------------------------------------------
__global__ __launch_bounds__(256)
void proj_kernel(const float* __restrict__ x,
                 const float* __restrict__ Wq, const float* __restrict__ Wk,
                 const float* __restrict__ Wv,
                 _Float16* __restrict__ Qo, _Float16* __restrict__ Ko,
                 _Float16* __restrict__ Vo)
{
  __shared__ _Float16 ldsX[128][32];    // x tile, row-major f16
  __shared__ _Float16 ldsWt[128][32];   // W tile transposed: [n][k]

  const float* Ws[3] = {Wq, Wk, Wv};
  _Float16*    Os[3] = {Qo, Ko, Vo};
  const float* Wsel = Ws[blockIdx.y];
  _Float16*    Osel = Os[blockIdx.y];

  const int row0  = blockIdx.x * 128;
  const int t     = threadIdx.x;
  const int wid   = t >> 5;
  const int lane  = t & 31;
  const int halfm = lane >> 4;
  const int ln    = lane & 15;
  const int kbase = halfm * 8;

  v8f acc[8] = {};

  for (int c0 = 0; c0 < C_; c0 += 32) {
    {  // stage x tile (f32 -> f16): thread t loads 16 contiguous floats
      int rr = t >> 1, seg = t & 1;
      const float* src = x + (long)(row0 + rr) * C_ + c0 + seg * 16;
#pragma unroll
      for (int i = 0; i < 16; ++i) ldsX[rr][seg * 16 + i] = (_Float16)src[i];
    }
    {  // stage W chunk transposed: W[c0+k][n] -> ldsWt[n][k]
      int k = t >> 3, nch = (t & 7) * 16;
      const float* src = Wsel + (long)(c0 + k) * H_ + nch;
#pragma unroll
      for (int i = 0; i < 16; ++i) ldsWt[nch + i][k] = (_Float16)src[i];
    }
    __syncthreads();

    v16h a = load_a16(&ldsX[wid * 16 + ln][0], kbase);
#pragma unroll
    for (int ch = 0; ch < 8; ++ch) {
      v16h b = load_a16(&ldsWt[ch * 16 + ln][0], kbase);
      acc[ch] = __builtin_amdgcn_wmma_f32_16x16x32_f16(
          false, a, false, b, (short)0, acc[ch], false, false);
    }
    __syncthreads();
  }

  // C/D layout: lanes 0-15 hold rows v, lanes 16-31 rows v+8; col = ln.
#pragma unroll
  for (int v = 0; v < 8; ++v) {
    const int  Mrow = v + halfm * 8;
    const long orow = (long)(row0 + wid * 16 + Mrow) * H_;
#pragma unroll
    for (int ch = 0; ch < 8; ++ch)
      Osel[orow + ch * 16 + ln] = (_Float16)acc[ch][v];
  }
}

// ---------------------------------------------------------------------------
// Flash attention per (cfg, b, s). Grid: (m/128, S, B), 256 threads = 8 waves.
// Wave owns 16 q rows. K and V tiles (32 dilated rows x 128 halves) are
// DMA'd into LDS by the Tensor Data Mover, double-buffered, while waves run
// WMMA + online softmax on the previous tile.
// ---------------------------------------------------------------------------
__global__ __launch_bounds__(256)
void attn_kernel(const _Float16* __restrict__ Q, const _Float16* __restrict__ K,
                 const _Float16* __restrict__ V,
                 float* __restrict__ O, float* __restrict__ Dn,
                 int w, int r, int S)
{
  __shared__ _Float16 bufK [2][32][H_];   // K tiles row-major  (TDM dest) 16 KB
  __shared__ _Float16 bufVs[2][32][H_];   // V tiles row-major  (TDM dest) 16 KB
  __shared__ _Float16 ldsVt[H_][32];      // V tile transposed [h][key]     8 KB
  __shared__ _Float16 ldsP [8][16][32];   // per-wave P relayout scratch    8 KB

  const int b     = blockIdx.z;
  const int s     = blockIdx.y;
  const int q0    = blockIdx.x * 128;
  const int t     = threadIdx.x;
  const int wid   = t >> 5;
  const int lane  = t & 31;
  const int halfm = lane >> 4;
  const int ln    = lane & 15;
  const int kbase = halfm * 8;

  const long segbase = (long)b * N_ + (long)s * w;      // row base in Q/K/V
  const unsigned long long kvstride = (unsigned long long)r * H_;

  // Q rows for this wave as A operands (4 chunks of 16x32 over h=128)
  const int qrow = q0 + wid * 16 + ln;                  // lane reads row M = ln
  const _Float16* qptr = Q + (segbase + (long)qrow * r) * H_;
  v16h qa[4];
#pragma unroll
  for (int c = 0; c < 4; ++c) qa[c] = load_a16(qptr + c * 32, kbase);

  v8f acc[8] = {};
  float mrow[8], lrow[8];
#pragma unroll
  for (int v = 0; v < 8; ++v) { mrow[v] = -1e30f; lrow[v] = 0.f; }

  const int   kend  = q0 + 128;                         // causal bound (<= M_)
  const float SCALE = 0.088388347648318447f;            // 1/sqrt(128)

  // prologue: DMA tile 0 into buffer 0 (wave 0 issues; TDM ignores EXEC)
  if (wid == 0) {
    tdm_load_2d_f16(lds_offset_of(&bufK [0][0][0]), K + segbase * H_, H_, 32, kvstride);
    tdm_load_2d_f16(lds_offset_of(&bufVs[0][0][0]), V + segbase * H_, H_, 32, kvstride);
  }

  int cur = 0;
  for (int k0 = 0; k0 < kend; k0 += 32, cur ^= 1) {
    const int nxt = cur ^ 1;
    if (k0 + 32 < kend) {                // prefetch next tile, then wait cur
      if (wid == 0) {
        const long nrow = segbase + (long)(k0 + 32) * r;
        tdm_load_2d_f16(lds_offset_of(&bufK [nxt][0][0]), K + nrow * H_, H_, 32, kvstride);
        tdm_load_2d_f16(lds_offset_of(&bufVs[nxt][0][0]), V + nrow * H_, H_, 32, kvstride);
      }
      __builtin_amdgcn_s_wait_tensorcnt(2);   // two newest outstanding => cur done
    } else {
      __builtin_amdgcn_s_wait_tensorcnt(0);
    }
    __syncthreads();                     // publish tile `cur` to all waves

    // V transpose, LDS->LDS (latency-tolerant; overlaps QK below)
    {
      int kr = t >> 3, chk = (t & 7) * 16;
      v8h v0 = *(const v8h*)&bufVs[cur][kr][chk];
      v8h v1 = *(const v8h*)&bufVs[cur][kr][chk + 8];
#pragma unroll
      for (int i = 0; i < 8; ++i) {
        ldsVt[chk + i][kr]     = v0[i];
        ldsVt[chk + 8 + i][kr] = v1[i];
      }
    }

    // S = Q K^T : two 16-key column tiles, accumulate over h in 32-chunks
    v8f sc0 = {}, sc1 = {};
#pragma unroll
    for (int c = 0; c < 4; ++c) {
      v16h bk0 = load_a16(&bufK[cur][ln][c * 32], kbase);
      v16h bk1 = load_a16(&bufK[cur][16 + ln][c * 32], kbase);
      sc0 = __builtin_amdgcn_wmma_f32_16x16x32_f16(
          false, qa[c], false, bk0, (short)0, sc0, false, false);
      sc1 = __builtin_amdgcn_wmma_f32_16x16x32_f16(
          false, qa[c], false, bk1, (short)0, sc1, false, false);
    }

    // online softmax over the 32 new columns
    float osc[8];
#pragma unroll
    for (int v = 0; v < 8; ++v) {
      const int  Mrow = v + halfm * 8;
      const int  qloc = q0 + wid * 16 + Mrow;
      float s0 = sc0[v] * SCALE, s1 = sc1[v] * SCALE;
      const bool msk0 = (k0 + ln)      > qloc;
      const bool msk1 = (k0 + 16 + ln) > qloc;
      if (msk0) s0 = -1e30f;
      if (msk1) s1 = -1e30f;
      float tmx = fmaxf(s0, s1);
#pragma unroll
      for (int off = 1; off < 16; off <<= 1)
        tmx = fmaxf(tmx, __shfl_xor(tmx, off, 32));
      const float newm = fmaxf(mrow[v], tmx);
      const float p0 = msk0 ? 0.f : __expf(s0 - newm);
      const float p1 = msk1 ? 0.f : __expf(s1 - newm);
      float ps = p0 + p1;
#pragma unroll
      for (int off = 1; off < 16; off <<= 1)
        ps += __shfl_xor(ps, off, 32);
      const float so = __expf(mrow[v] - newm);
      lrow[v] = lrow[v] * so + ps;
      mrow[v] = newm;
      osc[v]  = so;
      // C-layout -> A-layout relayout of P through per-wave LDS scratch
      ldsP[wid][Mrow][ln]      = (_Float16)p0;
      ldsP[wid][Mrow][16 + ln] = (_Float16)p1;
    }
#pragma unroll
    for (int ch = 0; ch < 8; ++ch)
#pragma unroll
      for (int v = 0; v < 8; ++v) acc[ch][v] *= osc[v];

    // make per-wave ldsP writes visible before the cross-lane re-read
    asm volatile("s_wait_dscnt 0x0" ::: "memory");

    __syncthreads();                     // ldsVt complete for all waves

    // O += P V : A = P (16x32), B^T rows come from ldsVt
    v16h pa = load_a16(&ldsP[wid][ln][0], kbase);
#pragma unroll
    for (int ch = 0; ch < 8; ++ch) {
      v16h bv = load_a16(&ldsVt[ch * 16 + ln][0], kbase);
      acc[ch] = __builtin_amdgcn_wmma_f32_16x16x32_f16(
          false, pa, false, bv, (short)0, acc[ch], false, false);
    }
  }

  // epilogue: normalized o and denom = l * exp(m) = exp(lse)
#pragma unroll
  for (int v = 0; v < 8; ++v) {
    const int   Mrow = v + halfm * 8;
    const int   qloc = q0 + wid * 16 + Mrow;
    const float inv  = 1.f / lrow[v];
    const long  obase = ((long)(b * S + s) * M_ + qloc) * H_;
#pragma unroll
    for (int ch = 0; ch < 8; ++ch)
      O[obase + ch * 16 + ln] = acc[ch][v] * inv;
    if (ln == 0)
      Dn[(long)(b * S + s) * M_ + qloc] = lrow[v] * __expf(mrow[v]);
  }
}

// ---------------------------------------------------------------------------
// Cross-config blend: out[b,n,:] = sum_cfg o_cfg * denom_cfg / denom_sum.
// Grid: B*N blocks of 128 threads (one per h).
// ---------------------------------------------------------------------------
__global__ __launch_bounds__(128)
void combine_kernel(const float* __restrict__ o0, const float* __restrict__ o1,
                    const float* __restrict__ o2, const float* __restrict__ o3,
                    const float* __restrict__ d0, const float* __restrict__ d1,
                    const float* __restrict__ d2, const float* __restrict__ d3,
                    float* __restrict__ out)
{
  const int bn = blockIdx.x;
  const int b  = bn >> 13;            // N_ = 8192
  const int n  = bn & (N_ - 1);
  const int h  = threadIdx.x;

  const int Wc[4] = {1024, 2048, 4096, 8192};
  const int Rc[4] = {1, 2, 4, 8};
  const int Sc[4] = {8, 4, 2, 1};
  const float* ob[4] = {o0, o1, o2, o3};
  const float* db[4] = {d0, d1, d2, d3};

  float dv[4]; long idx[4]; bool act[4];
  float dsum = 0.f;
#pragma unroll
  for (int i = 0; i < 4; ++i) {
    act[i] = (n & (Rc[i] - 1)) == 0;
    dv[i] = 0.f; idx[i] = 0;
    if (act[i]) {
      const int s = n / Wc[i];
      const int j = (n % Wc[i]) / Rc[i];
      idx[i] = (long)(b * Sc[i] + s) * M_ + j;
      dv[i]  = db[i][idx[i]];
      dsum  += dv[i];
    }
  }
  float o = 0.f;
#pragma unroll
  for (int i = 0; i < 4; ++i)
    if (act[i]) o += ob[i][idx[i] * H_ + h] * (dv[i] / dsum);
  out[(long)bn * H_ + h] = o;
}

// ---------------------------------------------------------------------------
extern "C" void kernel_launch(void* const* d_in, const int* in_sizes, int n_in,
                              void* d_out, int out_size, void* d_ws, size_t ws_size,
                              hipStream_t stream)
{
  (void)in_sizes; (void)n_in; (void)out_size; (void)ws_size;
  const float* x  = (const float*)d_in[0];
  const float* Wq = (const float*)d_in[1];
  const float* Wk = (const float*)d_in[2];
  const float* Wv = (const float*)d_in[3];
  float* out = (float*)d_out;

  char*  ws  = (char*)d_ws;
  size_t off = 0;
  auto alloc = [&](size_t bytes) {
    void* p = ws + off;
    off += (bytes + 255) & ~(size_t)255;
    return p;
  };

  const size_t rowsAll = (size_t)B_ * N_;
  _Float16* Qb = (_Float16*)alloc(rowsAll * H_ * sizeof(_Float16));
  _Float16* Kb = (_Float16*)alloc(rowsAll * H_ * sizeof(_Float16));
  _Float16* Vb = (_Float16*)alloc(rowsAll * H_ * sizeof(_Float16));

  const int Wc[4] = {1024, 2048, 4096, 8192};
  const int Rc[4] = {1, 2, 4, 8};
  const int Sc[4] = {8, 4, 2, 1};

  float* oB[4]; float* dB[4];
  for (int i = 0; i < 4; ++i) {
    const size_t rows = (size_t)B_ * (N_ / Rc[i]);   // = B*S*m
    oB[i] = (float*)alloc(rows * H_ * sizeof(float));
    dB[i] = (float*)alloc(rows * sizeof(float));
  }

  // 1) shared QKV projection (f32 -> f16), three GEMMs via blockIdx.y
  proj_kernel<<<dim3(rowsAll / 128, 3, 1), 256, 0, stream>>>(
      x, Wq, Wk, Wv, Qb, Kb, Vb);

  // 2) per-config flash attention (TDM-pipelined K/V)
  for (int i = 0; i < 4; ++i) {
    attn_kernel<<<dim3(M_ / 128, Sc[i], B_), 256, 0, stream>>>(
        Qb, Kb, Vb, oB[i], dB[i], Wc[i], Rc[i], Sc[i]);
  }

  // 3) cross-config denominator blend
  combine_kernel<<<dim3(B_ * N_), 128, 0, stream>>>(
      oB[0], oB[1], oB[2], oB[3], dB[0], dB[1], dB[2], dB[3], out);
}
